// QuantLinear_58978490909053
// MI455X (gfx1250) — 8-bit matrix operand order repaired and checked by reading, compiled, not run on hardware
//
#include <hip/hip_runtime.h>

#define M_DIM 8192
#define N_DIM 4096
#define K_DIM 4096

#define BM 128
#define BN 128
#define BK 64
#define KTILES (K_DIM / BK)

typedef __attribute__((ext_vector_type(8))) int   v8i;
typedef __attribute__((ext_vector_type(4))) int   v4i;
typedef __attribute__((ext_vector_type(2))) int   v2i;

__global__ __launch_bounds__(256) void quant_kernel(
    const float* __restrict__ x, const float* __restrict__ smooth,
    signed char* __restrict__ q, float* __restrict__ amax_out)
{
    const int row  = blockIdx.x;
    const int t    = threadIdx.x;
    const int lane = t & 31;
    const int wave = t >> 5;
    const float* xr = x + (size_t)row * K_DIM;

    float4 v[4];
    float am = 0.0f;
#pragma unroll
    for (int i = 0; i < 4; ++i) {
        const int idx = i * 1024 + t * 4;
        float4 xa = *(const float4*)(xr + idx);
        float4 sa = *(const float4*)(smooth + idx);
        v[i].x = xa.x * sa.x;  v[i].y = xa.y * sa.y;
        v[i].z = xa.z * sa.z;  v[i].w = xa.w * sa.w;
        am = fmaxf(am, fmaxf(fmaxf(fabsf(v[i].x), fabsf(v[i].y)),
                             fmaxf(fabsf(v[i].z), fabsf(v[i].w))));
    }

#pragma unroll
    for (int off = 16; off > 0; off >>= 1)
        am = fmaxf(am, __shfl_xor(am, off, 32));

    __shared__ float red[8];
    if (lane == 0) red[wave] = am;
    __syncthreads();
    am = red[0];
#pragma unroll
    for (int wv = 1; wv < 8; ++wv) am = fmaxf(am, red[wv]);

    const float scale = 127.0f / am;
    signed char* qr = q + (size_t)row * K_DIM;
#pragma unroll
    for (int i = 0; i < 4; ++i) {
        const int idx = i * 1024 + t * 4;
        float e[4] = { v[i].x, v[i].y, v[i].z, v[i].w };
        int packed = 0;
#pragma unroll
        for (int j = 0; j < 4; ++j) {
            float qf = floorf(e[j] * scale + 0.5f);
            qf = fminf(127.0f, fmaxf(-127.0f, qf));
            packed |= ((int)qf & 255) << (8 * j);
        }
        *(int*)(qr + idx) = packed;
    }
    if (t == 0) amax_out[row] = am;
}


#define GEMM_STEP(KT, BUF)                                                    \
  {                                                                           \
    const int k0_ = (KT) * BK;                                                \
    if ((KT) + 1 < KTILES) {                                                  \
      _Pragma("unroll")                                                       \
      for (int i = 0; i < 2; ++i) {                                           \
        const v4i av = *(const v4i*)(srcA[i] + k0_ + BK);                     \
        v2i lo; lo.x = av.x; lo.y = av.y;          \
        v2i hi; hi.x = av.z; hi.y = av.w;          \
        *(v2i*)(dstA[i][(BUF) ^ 1])       = lo;                               \
        *(v2i*)(dstA[i][(BUF) ^ 1] + 8) = hi;                                 \
      }                                                                       \
    }                                                                         \
    v8i af[4];                                                                \
    _Pragma("unroll")                                                         \
    for (int im = 0; im < 4; ++im)                                            \
      af[im] = *(const v8i*)(aLds[(BUF)] + im * 1024);                        \
    _Pragma("unroll")                                                         \
    for (int in = 0; in < 2; ++in) {                                          \
      const v4i blo = *(const v4i*)(wp[in] + k0_);                            \
      const v4i bhi = *(const v4i*)(wp[in] + k0_ + 32);                       \
      const v8i bf = { blo.x, blo.y, blo.z, blo.w,                            \
                       bhi.x, bhi.y, bhi.z, bhi.w };                          \
      _Pragma("unroll")                                                       \
      for (int im = 0; im < 4; ++im)                                          \
        acc[im][in] = __builtin_amdgcn_wmma_i32_16x16x64_iu8(                 \
             true, af[im],  true, bf,                       \
            acc[im][in],  false,  false);               \
    }                                                                         \
    __syncthreads();                                                          \
  }

__global__ __launch_bounds__(256) void gemm_kernel(
    const signed char* __restrict__ q, const signed char* __restrict__ w,
    const float* __restrict__ wq, const float* __restrict__ amax,
    float* __restrict__ out)
{
    __shared__ __align__(32) signed char lA[2][BM * BK];

    const int t    = threadIdx.x;
    const int lane = t & 31;
    const int wave = t >> 5;
    const int wm   = wave & 1;
    const int wn   = wave >> 1;
    const int block_m = blockIdx.y * BM;
    const int block_n = blockIdx.x * BN;

    const signed char* srcA[2];
    signed char*       dstA[2][2];
#pragma unroll
    for (int i = 0; i < 2; ++i) {
        const int c  = t + i * 256;
        const int r  = c >> 2;
        const int kc = c & 3;
        srcA[i] = q + (size_t)(block_m + r) * K_DIM + kc * 16;
        const int s  = r >> 4;
        const int m  = r & 15;
        const int vb = ((kc & 1) << 7) | ((kc >> 1) << 2);
        dstA[i][0] = &lA[0][s * 1024 + m * 32 + vb * 4];
        dstA[i][1] = &lA[1][s * 1024 + m * 32 + vb * 4];
    }

    const signed char* aLds[2];
    aLds[0] = &lA[0][wm * 4096 + lane * 32];
    aLds[1] = &lA[1][wm * 4096 + lane * 32];

    const signed char* wp[2];
#pragma unroll
    for (int in = 0; in < 2; ++in)
        wp[in] = w + (size_t)(block_n + wn * 32 + in * 16 + (lane & 15)) * K_DIM
                   + (lane >> 4) * 16;

    const v8i vzero = {0, 0, 0, 0, 0, 0, 0, 0};
    v8i acc[4][2];
#pragma unroll
    for (int im = 0; im < 4; ++im)
#pragma unroll
        for (int in = 0; in < 2; ++in) acc[im][in] = vzero;

#pragma unroll
    for (int i = 0; i < 2; ++i) {
        const v4i av = *(const v4i*)(srcA[i]);
        v2i lo; lo.x = av.x; lo.y = av.y;
        v2i hi; hi.x = av.z; hi.y = av.w;
        *(v2i*)(dstA[i][0])       = lo;
        *(v2i*)(dstA[i][0] + 8) = hi;
    }
    __syncthreads();

    for (int kt = 0; kt < KTILES; kt += 2) {
        GEMM_STEP(kt,     0)
        GEMM_STEP(kt + 1, 1)
    }

    const int nn = lane & 15;
    const int hh = lane >> 4;
    const float inv127 = 1.0f / 127.0f;

    float arow[4][8];
#pragma unroll
    for (int im = 0; im < 4; ++im)
#pragma unroll
        for (int j = 0; j < 8; ++j)
            arow[im][j] = amax[block_m + wm * 64 + im * 16 + hh * 8 + j];

#pragma unroll
    for (int in = 0; in < 2; ++in) {
        const int n  = block_n + wn * 32 + in * 16 + nn;
        const float sn = wq[n] * inv127;
#pragma unroll
        for (int im = 0; im < 4; ++im) {
            const int mbase = block_m + wm * 64 + im * 16 + hh * 8;
#pragma unroll
            for (int j = 0; j < 8; ++j)
                out[(size_t)(mbase + j) * N_DIM + n] =
                    (float)acc[im][in][j] * sn * arow[im][j];
        }
    }
}

extern "C" void kernel_launch(void* const* d_in, const int* in_sizes, int n_in,
                              void* d_out, int out_size, void* d_ws, size_t ws_size,
                              hipStream_t stream)
{
    const float*       x      = (const float*)d_in[0];
    const float*       smooth = (const float*)d_in[1];
    const float*       wq     = (const float*)d_in[2];
    const signed char* w      = (const signed char*)d_in[3];
    float*             out    = (float*)d_out;

    signed char* qbuf = (signed char*)d_ws;
    float*       amax = (float*)((char*)d_ws + (size_t)M_DIM * K_DIM);

    quant_kernel<<<M_DIM, 256, 0, stream>>>(x, smooth, qbuf, amax);

    dim3 grid(N_DIM / BN, M_DIM / BM);
    gemm_kernel<<<grid, 256, 0, stream>>>(qbuf, w, wq, amax, out);
}
